// MMDLoss_88390426952445
// MI455X (gfx1250) — matrix operand order repaired and checked by reading, not compiled, not run on hardware
//
#include <hip/hip_runtime.h>
#include <hip/hip_bf16.h>

// ---------------------------------------------------------------------------
// Energy-distance (geomloss) loss on MI455X / gfx1250.
//   loss = mean_b [ mean ||x-y|| - 0.5 mean ||x-x'|| - 0.5 mean ||y-y'|| ]
// Core = 3 batched GEMMs (512x512x128) + sqrt epilogue.
// f32 inputs are split into bf16 hi/lo planes so the dot products run on the
// bf16 WMMA pipe (V_WMMA_F32_16X16X32_BF16) with ~f32 accuracy:
//   dot(x,y) ~= hi.hi + hi.lo + lo.hi   (lo.lo term ~2^-16 relative, dropped)
// xx / yy distance matrices are symmetric: only upper-triangle tiles are
// computed; off-diagonal tiles carry double weight (exact, deterministic).
// ---------------------------------------------------------------------------

typedef __attribute__((ext_vector_type(16))) __bf16 v16bf;
typedef __attribute__((ext_vector_type(8)))  __bf16 v8bf;
typedef __attribute__((ext_vector_type(8)))  float  v8f;

#define NPTS 512   // points per cloud
#define DIM  128   // dimensions
#define TILE 64    // output tile edge per block

// --- LDS fragment loaders per CDNA5 ISA 7.12.2 (wave32) ---------------------
// A matrix 16x32 bf16: lanes 0-15 = rows, hold K {0..7,16..23}; lanes 16-31
// hold K {8..15,24..31} of the same rows.
__device__ inline v16bf load_a_frag(const __bf16* rowPtr, int lane, int kc) {
  const int k0 = kc * 32 + ((lane < 16) ? 0 : 8);
  const v8bf p0 = *(const v8bf*)(rowPtr + k0);        // K k0..k0+7   (16B)
  const v8bf p1 = *(const v8bf*)(rowPtr + k0 + 16);   // K k0+16..+23 (16B)
  return __builtin_shufflevector(p0, p1, 0, 1, 2, 3, 4, 5, 6, 7,
                                 8, 9, 10, 11, 12, 13, 14, 15);
}

// B matrix 32x16 bf16: lane%16 = column; lanes 0-15 hold K {0..7,16..23},
// lanes 16-31 hold K {8..15,24..31} (same K order as A).
__device__ inline v16bf load_b_frag(const __bf16* rowPtr, int lane, int kc) {
  const int k0 = kc * 32 + ((lane < 16) ? 0 : 8);
  const v8bf p0 = *(const v8bf*)(rowPtr + k0);
  const v8bf p1 = *(const v8bf*)(rowPtr + k0 + 16);
  return __builtin_shufflevector(p0, p1, 0, 1, 2, 3, 4, 5, 6, 7,
                                 8, 9, 10, 11, 12, 13, 14, 15);
}

__global__ __launch_bounds__(256) void mmd_tile_kernel(
    const float* __restrict__ x, const float* __restrict__ y,
    float* __restrict__ partials) {
  __shared__ __align__(32) __bf16 aHi[TILE][DIM];
  __shared__ __align__(32) __bf16 aLo[TILE][DIM];
  __shared__ __align__(32) __bf16 bHi[TILE][DIM];
  __shared__ __align__(32) __bf16 bLo[TILE][DIM];
  __shared__ float nA[TILE];
  __shared__ float nB[TILE];
  __shared__ float red[256];

  const int tid    = threadIdx.x;
  const int lane   = tid & 31;
  const int wave   = tid >> 5;
  const int tileId = blockIdx.x;   // 0..63 : (tm, tn) tile of the 512x512 matrix
  const int tm     = tileId >> 3;
  const int tn     = tileId & 7;
  const int b      = blockIdx.y;   // batch
  const int mode   = blockIdx.z;   // 0: x-y   1: x-x   2: y-y

  // Symmetric modes: skip strictly-lower-triangle tiles (mirrored by weight).
  if (mode != 0 && tm > tn) {
    if (tid == 0)
      partials[((size_t)mode * gridDim.y + b) * 64 + tileId] = 0.0f;
    return;
  }

  const float* Aptr = (mode == 2) ? y : x;
  const float* Bptr = (mode == 1) ? x : y;

  // ---- Stage tiles: global f32 -> (hi,lo) bf16 planes in LDS --------------
  const size_t aBase = ((size_t)b * NPTS + (size_t)tm * TILE) * DIM;
  const size_t bBase = ((size_t)b * NPTS + (size_t)tn * TILE) * DIM;
#pragma unroll
  for (int i = 0; i < 8; ++i) {           // 256 thr * 8 * float4 = 64x128 f32
    const int idx = tid + i * 256;
    const int row = idx >> 5;
    const int col = (idx & 31) << 2;
    const float4 va = *(const float4*)(Aptr + aBase + (size_t)row * DIM + col);
    const float4 vb = *(const float4*)(Bptr + bBase + (size_t)row * DIM + col);
    const float av[4] = {va.x, va.y, va.z, va.w};
    const float bv[4] = {vb.x, vb.y, vb.z, vb.w};
#pragma unroll
    for (int j = 0; j < 4; ++j) {
      const __bf16 ah = (__bf16)av[j];
      aHi[row][col + j] = ah;
      aLo[row][col + j] = (__bf16)(av[j] - (float)ah);
      const __bf16 bh = (__bf16)bv[j];
      bHi[row][col + j] = bh;
      bLo[row][col + j] = (__bf16)(bv[j] - (float)bh);
    }
  }
  __syncthreads();

  // ---- Row norms (consistent with the hi+lo representation) ---------------
  if (tid < TILE) {
    float s = 0.f;
#pragma unroll 4
    for (int k = 0; k < DIM; ++k) {
      const float v = (float)aHi[tid][k] + (float)aLo[tid][k];
      s += v * v;
    }
    nA[tid] = s;
  } else if (tid < 2 * TILE) {
    const int r = tid - TILE;
    float s = 0.f;
#pragma unroll 4
    for (int k = 0; k < DIM; ++k) {
      const float v = (float)bHi[r][k] + (float)bLo[r][k];
      s += v * v;
    }
    nB[r] = s;
  }
  __syncthreads();

  // ---- WMMA: each wave owns 2 subtiles sharing one B column block ---------
  // subtile s0 = wave (sm in {0,1}), s1 = wave+8 (sm+2), same sn -> reuse B.
  const int sm0 = wave >> 2;
  const int sn  = wave & 3;
  const int sm1 = sm0 + 2;
  const int aRow0 = sm0 * 16 + (lane & 15);
  const int aRow1 = sm1 * 16 + (lane & 15);
  const int bRow  = sn * 16 + (lane & 15);

  v8f c0 = {};
  v8f c1 = {};
#pragma unroll
  for (int kc = 0; kc < DIM / 32; ++kc) {
    const v16bf bh  = load_b_frag(&bHi[bRow][0], lane, kc);
    const v16bf bl  = load_b_frag(&bLo[bRow][0], lane, kc);
    const v16bf ah0 = load_a_frag(&aHi[aRow0][0], lane, kc);
    const v16bf al0 = load_a_frag(&aLo[aRow0][0], lane, kc);
    const v16bf ah1 = load_a_frag(&aHi[aRow1][0], lane, kc);
    const v16bf al1 = load_a_frag(&aLo[aRow1][0], lane, kc);
    c0 = __builtin_amdgcn_wmma_f32_16x16x32_bf16(false, ah0, false, bh, (short)0, c0, false, false);
    c0 = __builtin_amdgcn_wmma_f32_16x16x32_bf16(false, ah0, false, bl, (short)0, c0, false, false);
    c0 = __builtin_amdgcn_wmma_f32_16x16x32_bf16(false, al0, false, bh, (short)0, c0, false, false);
    c1 = __builtin_amdgcn_wmma_f32_16x16x32_bf16(false, ah1, false, bh, (short)0, c1, false, false);
    c1 = __builtin_amdgcn_wmma_f32_16x16x32_bf16(false, ah1, false, bl, (short)0, c1, false, false);
    c1 = __builtin_amdgcn_wmma_f32_16x16x32_bf16(false, al1, false, bh, (short)0, c1, false, false);
  }

  // ---- Epilogue: d = sqrt(max(|a|^2+|b|^2-2ab, 0)), accumulate ------------
  // C layout: lanes 0-15 -> M=v, lanes 16-31 -> M=v+8; N = lane%16.
  float lsum = 0.f;
  const int mOff = (lane >> 4) << 3;
  const int nIdx = sn * 16 + (lane & 15);
  const float bn = nB[nIdx];
#pragma unroll
  for (int v = 0; v < 8; ++v) {
    const float d2 = nA[sm0 * 16 + mOff + v] + bn - 2.0f * c0[v];
    lsum += (d2 > 0.f) ? __builtin_sqrtf(d2) : 0.f;
    const float e2 = nA[sm1 * 16 + mOff + v] + bn - 2.0f * c1[v];
    lsum += (e2 > 0.f) ? __builtin_sqrtf(e2) : 0.f;
  }

  // ---- Deterministic fixed-order block reduction --------------------------
  red[tid] = lsum;
  __syncthreads();
#pragma unroll
  for (int off = 128; off > 0; off >>= 1) {
    if (tid < off) red[tid] += red[tid + off];
    __syncthreads();
  }
  if (tid == 0) {
    // cross: +1; symmetric modes: -0.5 on diagonal tiles, -1.0 off-diagonal
    // (each upper tile also stands in for its mirrored lower tile).
    const float w = (mode == 0) ? 1.0f : ((tm == tn) ? -0.5f : -1.0f);
    partials[((size_t)mode * gridDim.y + b) * 64 + tileId] = w * red[0];
  }
}

__global__ __launch_bounds__(256) void mmd_reduce_kernel(
    const float* __restrict__ partials, int n, float scale,
    float* __restrict__ out) {
  __shared__ float red[256];
  float s = 0.f;
  for (int i = threadIdx.x; i < n; i += 256) s += partials[i];  // fixed order
  red[threadIdx.x] = s;
  __syncthreads();
#pragma unroll
  for (int off = 128; off > 0; off >>= 1) {
    if (threadIdx.x < off) red[threadIdx.x] += red[threadIdx.x + off];
    __syncthreads();
  }
  if (threadIdx.x == 0) out[0] = red[0] * scale;
}

extern "C" void kernel_launch(void* const* d_in, const int* in_sizes, int n_in,
                              void* d_out, int out_size, void* d_ws, size_t ws_size,
                              hipStream_t stream) {
  const float* x = (const float*)d_in[0];
  const float* y = (const float*)d_in[1];
  float* partials = (float*)d_ws;              // 3 * nBatch * 64 floats
  const int nBatch = in_sizes[0] / (NPTS * DIM);  // 128

  dim3 grid(64, nBatch, 3);                    // tiles x batches x modes
  mmd_tile_kernel<<<grid, 256, 0, stream>>>(x, y, partials);

  const int nPart = 3 * nBatch * 64;
  const float scale = 1.0f / ((float)nBatch * (float)NPTS * (float)NPTS);
  mmd_reduce_kernel<<<1, 256, 0, stream>>>(partials, nPart, scale, (float*)d_out);
}
